// ThreeLayerGCN_10204842295477
// MI455X (gfx1250) — matrix operand order repaired and checked by reading, not compiled, not run on hardware
//
#include <hip/hip_runtime.h>
#include <hip/hip_bf16.h>

typedef __attribute__((ext_vector_type(16))) _Float16 v16h;
typedef __attribute__((ext_vector_type(8)))  float    v8f;

#define GK 128   // inner dimension is 128 for all three layers

// ---------------------------------------------------------------------------
// Degree / normalization kernels
// ---------------------------------------------------------------------------
__global__ void k_deg_init(float* __restrict__ deg, int n) {
    int i = blockIdx.x * blockDim.x + threadIdx.x;
    if (i < n) deg[i] = 1.0f;                 // self-loop contributes 1
}

__global__ void k_deg_count(const long long* __restrict__ dst,
                            float* __restrict__ deg, int E) {
    int i = blockIdx.x * blockDim.x + threadIdx.x;
    if (i < E) atomicAdd(&deg[(int)dst[i]], 1.0f);
}

__global__ void k_dinv(float* __restrict__ deg, int n) {
    int i = blockIdx.x * blockDim.x + threadIdx.x;
    if (i < n) deg[i] = rsqrtf(deg[i]);       // deg >= 1 always
}

// ---------------------------------------------------------------------------
// WMMA GEMM:  H[M x C] = A[M x 128] * W[128 x C]   (fp16 in, fp32 acc)
//
// * One wave computes a full 16 x C row-strip (TN = C/16 tiles): each A
//   fragment is loaded once and reused across all TN WMMAs per k-step.
// * W is staged in LDS once per block, pre-converted to fp16 and pre-swizzled
//   into this kernel's B-fragment lane layout (contiguous k per half-wave), so each lane's v16h B fragment is
//   32 contiguous bytes in LDS (2x ds_load_b128 instead of 16 global loads).
//   LDS layout: Wsh[(kb*TN + tn)*32 + lane][e], e=0..15, where the half at
//   (lane, e) is W[kb*32 + (lane>>4)*16 + e][tn*16 + (lane&15)].
// * RELU (applied to A on read) is a compile-time flag -> branch-free.
// M = 50000 = 3125*16 exactly; C in {128, 64}.
// ---------------------------------------------------------------------------
template <int TN, bool RELU>
__global__ void k_gemm_wmma(const float* __restrict__ A,
                            const float* __restrict__ W,
                            float* __restrict__ H, int M) {
    constexpr int C = TN * 16;
    __shared__ alignas(32) _Float16 Wsh[4 * TN * 32 * 16];   // 4 k-blocks

    // ---- cooperative staging of W: coalesced fp32 read, swizzled fp16 store
    for (int i = threadIdx.x; i < GK * C; i += blockDim.x) {
        int k   = i / C;            // 0..127
        int c   = i - k * C;        // 0..C-1
        int kb  = k >> 5;           // k-block (32 wide)
        int hsk = (k >> 4) & 1;     // which half-wave holds this K
        int e   = k & 15;           // position within lane's 16 halves
        int tn  = c >> 4;
        int L   = hsk * 16 + (c & 15);
        Wsh[(((kb * TN + tn) * 32) + L) * 16 + e] = (_Float16)W[i];
    }
    __syncthreads();

    const int lane = threadIdx.x & 31;
    const int wave = blockIdx.x * (blockDim.x >> 5) + (threadIdx.x >> 5);
    if (wave >= (M >> 4)) return;             // wave-uniform: EXEC stays all-1s

    const int idx = lane & 15;                // M-row for A, N-col for B/D
    const int hs  = lane >> 4;                // half-wave select

    const float* arow = A + (size_t)(wave * 16 + idx) * GK;

    v8f zero = {};
    v8f acc[TN];
#pragma unroll
    for (int t = 0; t < TN; ++t) acc[t] = zero;

#pragma unroll
    for (int kb = 0; kb < 4; ++kb) {
        const int k0 = kb * 32;
        // ---- A fragment: 16x32 fp16, contiguous-k layout (lane=M, hs offsets K by 16) ----
        v16h a;
#pragma unroll
        for (int j = 0; j < 4; ++j) {
            float x0 = arow[k0 +      hs * 16 + 2 * j];
            float x1 = arow[k0 +      hs * 16 + 2 * j + 1];
            float x2 = arow[k0 + 8 + hs * 16 + 2 * j];
            float x3 = arow[k0 + 8 + hs * 16 + 2 * j + 1];
            if (RELU) {
                x0 = fmaxf(x0, 0.0f); x1 = fmaxf(x1, 0.0f);
                x2 = fmaxf(x2, 0.0f); x3 = fmaxf(x3, 0.0f);
            }
            a[2 * j]         = (_Float16)x0;
            a[2 * j + 1]     = (_Float16)x1;
            a[8 + 2 * j]     = (_Float16)x2;
            a[8 + 2 * j + 1] = (_Float16)x3;
        }
        // ---- sweep all N-tiles with this A fragment; B comes from LDS ----
#pragma unroll
        for (int tn = 0; tn < TN; ++tn) {
            const v16h b =
                *(const v16h*)&Wsh[(((kb * TN + tn) * 32) + lane) * 16];
            acc[tn] = __builtin_amdgcn_wmma_f32_16x16x32_f16(
                false, a, false, b, (short)0, acc[tn], false, false);
        }
    }

    // ---- D store: lane=N, VGPR r holds M = r + 8*hs ----
#pragma unroll
    for (int tn = 0; tn < TN; ++tn) {
        float* hbase = H + (size_t)(wave * 16) * C + tn * 16 + idx;
#pragma unroll
        for (int r = 0; r < 8; ++r) {
            hbase[(size_t)(r + 8 * hs) * C] = acc[tn][r];
        }
    }
}

// ---------------------------------------------------------------------------
// OUT[n][c] = H[n][c] * dinv[n]^2 + bias[c]   (self-loop term + bias)
// ---------------------------------------------------------------------------
__global__ void k_init_out(const float* __restrict__ H,
                           const float* __restrict__ dinv,
                           const float* __restrict__ bias,
                           float* __restrict__ OUT, int N, int C) {
    int t = blockIdx.x * blockDim.x + threadIdx.x;
    int perRow = C >> 2;
    if (t >= N * perRow) return;
    int r  = t / perRow;
    int c4 = (t - r * perRow) << 2;
    float dd = dinv[r]; dd *= dd;
    const float4 h = *(const float4*)(H + (size_t)r * C + c4);
    float4 o;
    o.x = h.x * dd + bias[c4 + 0];
    o.y = h.y * dd + bias[c4 + 1];
    o.z = h.z * dd + bias[c4 + 2];
    o.w = h.w * dd + bias[c4 + 3];
    *(float4*)(OUT + (size_t)r * C + c4) = o;
}

// ---------------------------------------------------------------------------
// Edge scatter: one wave per edge.  OUT[dst] += H[src] * dinv[src]*dinv[dst]
// Feature rows are L2-resident (25.6 MB << 192 MB L2).
// ---------------------------------------------------------------------------
__global__ void k_scatter(const long long* __restrict__ src,
                          const long long* __restrict__ dst,
                          const float* __restrict__ dinv,
                          const float* __restrict__ H,
                          float* __restrict__ OUT, int E, int C) {
    int wave = (blockIdx.x * blockDim.x + threadIdx.x) >> 5;
    int lane = threadIdx.x & 31;
    if (wave >= E) return;
    int s = (int)src[wave];
    int d = (int)dst[wave];
    float nrm = dinv[s] * dinv[d];
    const float* hs = H + (size_t)s * C;
    float*       od = OUT + (size_t)d * C;
    for (int c = lane; c < C; c += 32)
        atomicAdd(&od[c], hs[c] * nrm);
}

// ---------------------------------------------------------------------------
// Host-side orchestration
// ---------------------------------------------------------------------------
extern "C" void kernel_launch(void* const* d_in, const int* in_sizes, int n_in,
                              void* d_out, int out_size, void* d_ws, size_t ws_size,
                              hipStream_t stream) {
    const float*     x   = (const float*)d_in[0];
    const long long* ei  = (const long long*)d_in[1];   // int64 edge_index [2, E]
    const float*     W1  = (const float*)d_in[2];
    const float*     b1  = (const float*)d_in[3];
    const float*     W2  = (const float*)d_in[4];
    const float*     b2  = (const float*)d_in[5];
    const float*     W3  = (const float*)d_in[6];
    const float*     b3  = (const float*)d_in[7];
    float*           out = (float*)d_out;

    const int N = in_sizes[0] / 128;     // 50000
    const int E = in_sizes[1] / 2;       // 800000

    const long long* srcp = ei;
    const long long* dstp = ei + E;

    // workspace layout: dinv | bufH (N*128 f) | bufO (N*128 f)
    char*  ws    = (char*)d_ws;
    float* dinv  = (float*)ws;
    size_t off0  = ((size_t)N * sizeof(float) + 255) & ~(size_t)255;
    float* bufH  = (float*)(ws + off0);
    float* bufO  = (float*)(ws + off0 + (size_t)N * 128 * sizeof(float));

    const int TB = 256;

    // ---- degree -> dinv ----
    k_deg_init <<<(N + TB - 1) / TB, TB, 0, stream>>>(dinv, N);
    k_deg_count<<<(E + TB - 1) / TB, TB, 0, stream>>>(dstp, dinv, E);
    k_dinv     <<<(N + TB - 1) / TB, TB, 0, stream>>>(dinv, N);

    const int strips  = N >> 4;                 // 3125 row-strips
    const int wavesB  = TB >> 5;                // 8 waves / block
    const int gblocks = (strips + wavesB - 1) / wavesB;

    auto aggregate = [&](const float* H, const float* bias, float* O, int C) {
        int tot = N * (C >> 2);
        k_init_out<<<(tot + TB - 1) / TB, TB, 0, stream>>>(H, dinv, bias, O, N, C);
        int thr = E * 32;
        k_scatter<<<(thr + TB - 1) / TB, TB, 0, stream>>>(srcp, dstp, dinv, H, O, E, C);
    };

    // ---- layer 1: H1 = x @ W1 ----
    k_gemm_wmma<8, false><<<gblocks, TB, 0, stream>>>(x, W1, bufH, N);
    aggregate(bufH, b1, bufO, 128);             // bufO = pre-ReLU out1
    // ---- layer 2: H2 = relu(out1) @ W2 (ReLU fused into A read) ----
    k_gemm_wmma<8, true><<<gblocks, TB, 0, stream>>>(bufO, W2, bufH, N);
    aggregate(bufH, b2, bufO, 128);             // bufO = pre-ReLU out2
    // ---- layer 3: H3 = relu(out2) @ W3 ----
    k_gemm_wmma<4, true><<<gblocks, TB, 0, stream>>>(bufO, W3, bufH, N);
    aggregate(bufH, b3, out, 64);               // final output, no ReLU
}
